// LSTMDecoder_63702954935058
// MI455X (gfx1250) — compile-verified, matrix operand order checked by reading, not run on hardware
//
#include <hip/hip_runtime.h>
#include <hip/hip_bf16.h>

#define B_ 32
#define T_ 32
#define S_ 400
#define H_ 1024
#define I_ 1024
#define G4 4096

typedef __attribute__((ext_vector_type(16))) __bf16 v16bf;
typedef __attribute__((ext_vector_type(8)))  float  v8f;

// ---- WMMA fragment packing (CDNA5 ISA 7.12.2, 16-bit 16x32 operand layout) ----
// lane L: row/col = L%16 ; k-slots = {0..7,16..23} for L<16, {8..15,24..31} for L>=16
__device__ __forceinline__ v16bf pack_f32(const float* row, int k0, int hs) {
  v16bf v;
#pragma unroll
  for (int g = 0; g < 2; ++g)
#pragma unroll
    for (int j = 0; j < 8; ++j)
      v[g * 8 + j] = (__bf16)row[k0 + g * 16 + hs * 8 + j];
  return v;
}
__device__ __forceinline__ v16bf pack_bf16(const __bf16* row, int k0, int hs) {
  v16bf v;
#pragma unroll
  for (int g = 0; g < 2; ++g)
#pragma unroll
    for (int j = 0; j < 8; ++j)
      v[g * 8 + j] = row[k0 + g * 16 + hs * 8 + j];
  return v;
}
__device__ __forceinline__ v8f wmma_bf16(v16bf a, v16bf b, v8f c) {
  return __builtin_amdgcn_wmma_f32_16x16x32_bf16(false, a, false, b, (short)0, c,
                                                 false, false);
}
__device__ __forceinline__ float sigm(float x) { return 1.f / (1.f + __expf(-x)); }

// ---- one-time: convert weights + input to bf16, fuse biases ----
__global__ void k_convert(const float* Wih, const float* Whh, const float* Wout,
                          const float* Wattn, const float* x, const float* bih,
                          const float* bhh, __bf16* Wih_bf, __bf16* Whh_bf,
                          __bf16* Wout_bf, __bf16* Wattn_bf, __bf16* x_bf,
                          float* bias) {
  const long NIH = (long)G4 * 2048, NHH = (long)G4 * H_, NOUT = (long)H_ * 2048;
  const long NATTN = (long)H_ * H_, NX = (long)B_ * T_ * I_;
  const long E1 = NIH, E2 = E1 + NHH, E3 = E2 + NOUT, E4 = E3 + NATTN, E5 = E4 + NX;
  const long total = E5 + G4;
  for (long i = blockIdx.x * (long)blockDim.x + threadIdx.x; i < total;
       i += (long)gridDim.x * blockDim.x) {
    if (i < E1)      Wih_bf[i]        = (__bf16)Wih[i];
    else if (i < E2) Whh_bf[i - E1]   = (__bf16)Whh[i - E1];
    else if (i < E3) Wout_bf[i - E2]  = (__bf16)Wout[i - E2];
    else if (i < E4) Wattn_bf[i - E3] = (__bf16)Wattn[i - E3];
    else if (i < E5) x_bf[i - E4]     = (__bf16)x[i - E4];
    else { long j = i - E5; bias[j] = bih[j] + bhh[j]; }
  }
}

// ---- one-time: seed f32 state + bf16 mirrors ----
__global__ void k_init(const float* h0, const float* c0, const float* ha0, float* h,
                       float* c, float* hattn, __bf16* h_bf, __bf16* hattn_bf) {
  int idx = blockIdx.x * blockDim.x + threadIdx.x;
  if (idx >= B_ * H_) return;
  float hv = h0[idx], av = ha0[idx];
  h[idx] = hv;           c[idx] = c0[idx];       hattn[idx] = av;
  h_bf[idx] = (__bf16)hv; hattn_bf[idx] = (__bf16)av;
}

// ---- hoisted: enhy_new[b,s,g] = sum_h enc[s,b,h]*Wattn[g,h] (M=12800,N=1024,K=1024)
// 4 N-tiles per wave: encoder A-fragment fetched 16x instead of 64x.
__global__ void __launch_bounds__(256) k_enhy(const float* enc, const __bf16* Wattn_bf,
                                              __bf16* enhy_bf) {
  int wave = (blockIdx.x * blockDim.x + threadIdx.x) >> 5;   // 0..12799
  int lane = threadIdx.x & 31;
  int tm = wave >> 4, tng = wave & 15;
  int hs = lane >> 4, rl = lane & 15;
  int row = tm * 16 + rl;            // b*S + s
  int b = row / S_, s = row % S_;
  const float* arow = enc + ((long)s * B_ + b) * H_;
  int n[4];
  const __bf16* brow[4];
#pragma unroll
  for (int j = 0; j < 4; ++j) {
    n[j] = (tng * 4 + j) * 16 + rl;
    brow[j] = Wattn_bf + (long)n[j] * H_;
  }
  v8f acc[4] = {{}, {}, {}, {}};
  for (int k0 = 0; k0 < H_; k0 += 32) {
    v16bf a = pack_f32(arow, k0, hs);
#pragma unroll
    for (int j = 0; j < 4; ++j)
      acc[j] = wmma_bf16(a, pack_bf16(brow[j], k0, hs), acc[j]);
  }
#pragma unroll
  for (int j = 0; j < 4; ++j)
#pragma unroll
    for (int r = 0; r < 8; ++r) {
      int rg = tm * 16 + r + 8 * hs;
      enhy_bf[(long)rg * H_ + n[j]] = (__bf16)acc[j][r];
    }
}

// ---- gates = [x_t,h_attn]@Wih^T + h@Whh^T + bias  (M=32,N=4096,K=3072)
// Both M-tiles per wave: each weight fragment loaded once per step.
__global__ void __launch_bounds__(128) k_gates(const __bf16* x_bf, const __bf16* h_bf,
                                               const __bf16* hattn_bf,
                                               const __bf16* Wih_bf,
                                               const __bf16* Whh_bf, const float* bias,
                                               float* gates, int t) {
  int wave = (blockIdx.x * blockDim.x + threadIdx.x) >> 5;   // 0..255 = N tile
  int lane = threadIdx.x & 31;
  int hs = lane >> 4, rl = lane & 15;
  int n = wave * 16 + rl;
  int b0 = rl, b1 = 16 + rl;
  v8f acc0 = {}, acc1 = {};
  for (int k0 = 0; k0 < 3072; k0 += 32) {
    const __bf16 *a0, *a1; int ak;
    if (k0 < 1024) {
      a0 = x_bf + ((long)b0 * T_ + t) * I_; a1 = x_bf + ((long)b1 * T_ + t) * I_;
      ak = k0;
    } else if (k0 < 2048) {
      a0 = hattn_bf + (long)b0 * H_; a1 = hattn_bf + (long)b1 * H_; ak = k0 - 1024;
    } else {
      a0 = h_bf + (long)b0 * H_; a1 = h_bf + (long)b1 * H_; ak = k0 - 2048;
    }
    const __bf16* brw; int bk;
    if (k0 < 2048) { brw = Wih_bf + (long)n * 2048; bk = k0; }
    else           { brw = Whh_bf + (long)n * 1024; bk = k0 - 2048; }
    v16bf bb = pack_bf16(brw, bk, hs);
    acc0 = wmma_bf16(pack_bf16(a0, ak, hs), bb, acc0);
    acc1 = wmma_bf16(pack_bf16(a1, ak, hs), bb, acc1);
  }
  float bv = bias[n];
#pragma unroll
  for (int r = 0; r < 8; ++r) {
    int m = r + 8 * hs;
    gates[(long)m * G4 + n] = acc0[r] + bv;
    gates[(long)(16 + m) * G4 + n] = acc1[r] + bv;
  }
}

// ---- LSTM cell elementwise (writes f32 + bf16 mirror of h) ----
__global__ void k_cell(const float* gates, float* c, float* h, __bf16* h_bf) {
  int idx = blockIdx.x * blockDim.x + threadIdx.x;
  if (idx >= B_ * H_) return;
  int b = idx >> 10, n = idx & 1023;
  const float* g = gates + (long)b * G4;
  float iv = sigm(g[n]);
  float fv = sigm(g[H_ + n]);
  float gv = tanhf(g[2 * H_ + n]);
  float ov = sigm(g[3 * H_ + n]);
  float cv = fv * c[idx] + iv * gv;
  float hv = ov * tanhf(cv);
  c[idx] = cv;
  h[idx] = hv;
  h_bf[idx] = (__bf16)hv;
}

// ---- scores + softmax:  attn[b,s] = softmax_s( enhy_new[b,s,:] . h[b,:] ) ----
__global__ void __launch_bounds__(512) k_attn(const __bf16* enhy_bf, const float* h,
                                              float* attnw, float* out_attns, int t) {
  __shared__ float sc[S_];
  __shared__ float sred[2];
  int b = blockIdx.x;
  int lane = threadIdx.x & 31;
  int wave = threadIdx.x >> 5;   // 0..15
  const float* hr = h + (long)b * H_;
  for (int s = wave; s < S_; s += 16) {
    const __bf16* er = enhy_bf + ((long)b * S_ + s) * H_;
    float p = 0.f;
    for (int k = lane; k < H_; k += 32) p += (float)er[k] * hr[k];
#pragma unroll
    for (int o = 16; o; o >>= 1) p += __shfl_xor(p, o, 32);
    if (lane == 0) sc[s] = p;
  }
  __syncthreads();
  if (wave == 0) {
    float m = -1e30f;
    for (int s = lane; s < S_; s += 32) m = fmaxf(m, sc[s]);
#pragma unroll
    for (int o = 16; o; o >>= 1) m = fmaxf(m, __shfl_xor(m, o, 32));
    if (lane == 0) sred[0] = m;
  }
  __syncthreads();
  float mx = sred[0];
  for (int s = threadIdx.x; s < S_; s += blockDim.x) sc[s] = __expf(sc[s] - mx);
  __syncthreads();
  if (wave == 0) {
    float sm = 0.f;
    for (int s = lane; s < S_; s += 32) sm += sc[s];
#pragma unroll
    for (int o = 16; o; o >>= 1) sm += __shfl_xor(sm, o, 32);
    if (lane == 0) sred[1] = sm;
  }
  __syncthreads();
  float inv = 1.f / sred[1];
  for (int s = threadIdx.x; s < S_; s += blockDim.x) {
    float a = sc[s] * inv;
    attnw[(long)b * S_ + s] = a;
    out_attns[((long)t * B_ + b) * S_ + s] = a;
  }
}

// ---- ctx[b,h] = sum_s attn[b,s] * enc[s,b,h]  (writes bf16 for the out-GEMM) ----
__global__ void __launch_bounds__(256) k_ctx(const float* enc, const float* attnw,
                                             __bf16* ctx_bf) {
  __shared__ float sa[S_];
  int b = blockIdx.x;
  for (int s = threadIdx.x; s < S_; s += 256) sa[s] = attnw[(long)b * S_ + s];
  __syncthreads();
  int h0 = threadIdx.x * 4;
  float4 acc = {0.f, 0.f, 0.f, 0.f};
  for (int s = 0; s < S_; ++s) {
    float a = sa[s];
    const float4 e = *(const float4*)(enc + ((long)s * B_ + b) * H_ + h0);
    acc.x += a * e.x; acc.y += a * e.y; acc.z += a * e.z; acc.w += a * e.w;
  }
  __bf16* d = ctx_bf + (long)b * H_ + h0;
  d[0] = (__bf16)acc.x; d[1] = (__bf16)acc.y; d[2] = (__bf16)acc.z; d[3] = (__bf16)acc.w;
}

// ---- h_attn = tanh([ctx,h]@Wout^T)  (M=32,N=1024,K=2048), fused tanh + stores ----
__global__ void __launch_bounds__(128) k_out(const __bf16* ctx_bf, const __bf16* h_bf,
                                             const __bf16* Wout_bf, float* hattn,
                                             __bf16* hattn_bf, float* out_output,
                                             int t) {
  int wave = (blockIdx.x * blockDim.x + threadIdx.x) >> 5;   // 0..127
  int lane = threadIdx.x & 31;
  int tm = wave >> 6, tn = wave & 63;
  int hs = lane >> 4, rl = lane & 15;
  int brow = tm * 16 + rl;
  int n = tn * 16 + rl;
  v8f acc = {};
  for (int k0 = 0; k0 < 2048; k0 += 32) {
    const __bf16* arow; int ak;
    if (k0 < 1024) { arow = ctx_bf + (long)brow * H_; ak = k0; }
    else           { arow = h_bf + (long)brow * H_;   ak = k0 - 1024; }
    acc = wmma_bf16(pack_bf16(arow, ak, hs),
                    pack_bf16(Wout_bf + (long)n * 2048, k0, hs), acc);
  }
#pragma unroll
  for (int r = 0; r < 8; ++r) {
    int m = r + 8 * hs;
    int b = tm * 16 + m;
    float v = tanhf(acc[r]);
    hattn[(long)b * H_ + n] = v;
    hattn_bf[(long)b * H_ + n] = (__bf16)v;
    out_output[((long)b * T_ + t) * H_ + n] = v;
  }
}

// ---- final: h, c, h_attn, past_attn, loss ----
__global__ void k_final(const float* h, const float* c, const float* hattn,
                        const float* past, float* out) {
  const long OFF_H = (long)B_ * T_ * H_;                              // 1048576
  const long OFF_C = OFF_H + (long)B_ * H_;
  const long OFF_HA = OFF_C + (long)B_ * H_;
  const long OFF_PAST = OFF_HA + (long)B_ * H_ + (long)T_ * B_ * S_;  // 1556480
  const long OFF_LOSS = OFF_PAST + (long)B_ * S_;                     // 1569280
  int idx = blockIdx.x * blockDim.x + threadIdx.x;
  if (idx < B_ * H_) {
    out[OFF_H + idx] = h[idx];
    out[OFF_C + idx] = c[idx];
    out[OFF_HA + idx] = hattn[idx];
  }
  if (idx < B_ * S_) out[OFF_PAST + idx] = past[idx];
  if (idx == 0) out[OFF_LOSS] = 0.0f;
}

extern "C" void kernel_launch(void* const* d_in, const int* in_sizes, int n_in,
                              void* d_out, int out_size, void* d_ws, size_t ws_size,
                              hipStream_t stream) {
  const float* input_ = (const float*)d_in[0];
  const float* h0     = (const float*)d_in[1];
  const float* c0     = (const float*)d_in[2];
  const float* ha0    = (const float*)d_in[3];
  const float* enc    = (const float*)d_in[4];
  const float* past   = (const float*)d_in[5];
  const float* Wih    = (const float*)d_in[6];
  const float* Whh    = (const float*)d_in[7];
  const float* bih    = (const float*)d_in[8];
  const float* bhh    = (const float*)d_in[9];
  const float* Wattn  = (const float*)d_in[10];
  const float* Wout   = (const float*)d_in[11];

  char* ws = (char*)d_ws;
  __bf16* Wih_bf   = (__bf16*)(ws + 0);          // 16 MB
  __bf16* Whh_bf   = (__bf16*)(ws + 16777216);   //  8 MB
  __bf16* Wout_bf  = (__bf16*)(ws + 25165824);   //  4 MB
  __bf16* Wattn_bf = (__bf16*)(ws + 29360128);   //  2 MB
  float*  bias     = (float*) (ws + 31457280);   // 16 KB
  __bf16* x_bf     = (__bf16*)(ws + 31473664);   //  2 MB
  __bf16* enhy_bf  = (__bf16*)(ws + 33570816);   // 25 MB
  float*  h        = (float*) (ws + 59785216);
  float*  c        = (float*) (ws + 59916288);
  float*  hattn    = (float*) (ws + 60047360);
  __bf16* h_bf     = (__bf16*)(ws + 60178432);
  __bf16* hattn_bf = (__bf16*)(ws + 60243968);
  __bf16* ctx_bf   = (__bf16*)(ws + 60309504);
  float*  gates    = (float*) (ws + 60375040);
  float*  attnw    = (float*) (ws + 60899328);

  float* out = (float*)d_out;
  float* out_attns = out + 1146880;   // B*T*H + 3*B*H

  k_convert<<<4096, 256, 0, stream>>>(Wih, Whh, Wout, Wattn, input_, bih, bhh, Wih_bf,
                                      Whh_bf, Wout_bf, Wattn_bf, x_bf, bias);
  k_init<<<128, 256, 0, stream>>>(h0, c0, ha0, h, c, hattn, h_bf, hattn_bf);
  k_enhy<<<1600, 256, 0, stream>>>(enc, Wattn_bf, enhy_bf);

  for (int t = 0; t < T_; ++t) {
    k_gates<<<64, 128, 0, stream>>>(x_bf, h_bf, hattn_bf, Wih_bf, Whh_bf, bias, gates,
                                    t);
    k_cell<<<128, 256, 0, stream>>>(gates, c, h, h_bf);
    k_attn<<<32, 512, 0, stream>>>(enhy_bf, h, attnw, out_attns, t);
    k_ctx<<<32, 256, 0, stream>>>(enc, attnw, ctx_bf);
    k_out<<<32, 128, 0, stream>>>(ctx_bf, h_bf, Wout_bf, hattn, hattn_bf, out, t);
  }
  k_final<<<128, 256, 0, stream>>>(h, c, hattn, past, out);
}
